// InstrumentedBivectorBlock_37082747633887
// MI455X (gfx1250) — matrix operand order repaired and checked by reading, not compiled, not run on hardware
//
#include <hip/hip_runtime.h>
#include <math.h>

// ---------------------------------------------------------------- constants
#define BB    2
#define LL    1024
#define DD    256
#define BLTOK (BB * LL)      // 2048 tokens
#define NSETS 4
#define PPSN  4
#define NPL   16             // TOTAL_PLANES == POS_PLANES == 16
#define PI_F  3.14159265358979323846f

// ---------------------------------------------------------------- types
typedef __attribute__((ext_vector_type(16))) __bf16 v16bf;
typedef __attribute__((ext_vector_type(8)))  float  v8f;
typedef unsigned short u16;   // bf16 storage

union V16U { v16bf v; u16 s[16]; uint4 q[2]; };

__device__ __forceinline__ float bf2f(u16 s) {
    union { unsigned u; float f; } z; z.u = ((unsigned)s) << 16; return z.f;
}
__device__ __forceinline__ u16 f2bf(float f) {
    union { float f; unsigned u; } z; z.f = f;
    unsigned r = z.u + 0x7fffu + ((z.u >> 16) & 1u);   // round-to-nearest-even
    return (u16)(r >> 16);
}
__device__ __forceinline__ float wave_sum(float v) {
#pragma unroll
    for (int o = 16; o > 0; o >>= 1) v += __shfl_down(v, o, 32);
    return v;
}
__device__ __forceinline__ float sigm(float x) { return 1.0f / (1.0f + expf(-x)); }
__device__ __forceinline__ float gelu_exact(float x) {
    return 0.5f * x * (1.0f + erff(x * 0.70710678118654752f));
}
__device__ __forceinline__ float2 cmul(float2 a, float2 b) {
    return make_float2(a.x * b.x - a.y * b.y, a.x * b.y + a.y * b.x);
}

// ---------------------------------------------------------------- prep kernels
__global__ void cast_f32_to_bf16(const float* __restrict__ src, u16* __restrict__ dst, int n) {
    int i = blockIdx.x * blockDim.x + threadIdx.x;
    if (i < n) dst[i] = f2bf(src[i]);
}

// Wt[n*din + k] = W[k*dout + n]   (weight stored (din,dout) row-major -> (dout,din) bf16)
__global__ void transpose_cast(const float* __restrict__ W, u16* __restrict__ Wt,
                               int din, int dout) {
    int i = blockIdx.x * blockDim.x + threadIdx.x;
    if (i < din * dout) {
        int n = i / din, k = i % din;
        Wt[i] = f2bf(W[(size_t)k * dout + n]);
    }
}

// ---------------------------------------------------------------- WMMA GEMM
// C[M,N] = epilogue( A_bf16[M,K] @ Wt_bf16[N,K]^T + bias[N] )
#define MODE_F32        0
#define MODE_GELU_BF16  1
#define MODE_TANHPI_F32 2
#define MODE_RESID_F32  3

__device__ __forceinline__ v16bf load_a_frag(const u16* __restrict__ A, int K,
                                             int m0, int k0, int lane) {
    int half = lane >> 4, r = lane & 15;
    const u16* p = A + (size_t)(m0 + r) * K + k0 + 8 * half;
    V16U f;
    f.q[0] = *(const uint4*)(p);        // K = k0 + 8*half + [0..7]
    f.q[1] = *(const uint4*)(p + 16);   // K = k0 + 16 + 8*half + [0..7]
    return f.v;
}
__device__ __forceinline__ v16bf load_b_frag(const u16* __restrict__ Bt, int K,
                                             int n0, int k0, int lane) {
    int half = lane >> 4, c = lane & 15;
    const u16* p = Bt + (size_t)(n0 + c) * K + k0 + 8 * half;
    V16U f;
    f.q[0] = *(const uint4*)(p);        // K = k0 + 8*half + [0..7]
    f.q[1] = *(const uint4*)(p + 16);   // K = k0 + 16 + 8*half + [0..7]
    return f.v;
}

__global__ __launch_bounds__(128) void wmma_gemm(
    const u16* __restrict__ A, const u16* __restrict__ Bt,
    const float* __restrict__ bias, const float* __restrict__ resid,
    float* __restrict__ Cf, u16* __restrict__ Cb,
    int M, int N, int K, int mode)
{
    int lane = threadIdx.x & 31;
    int wave = threadIdx.x >> 5;
    int m0 = blockIdx.x * 16;
    int n0 = (blockIdx.y * 4 + wave) * 16;
    if (m0 >= M || n0 >= N) return;

    v8f acc = {};
    for (int k0 = 0; k0 < K; k0 += 32) {
        v16bf a = load_a_frag(A, K, m0, k0, lane);
        v16bf b = load_b_frag(Bt, K, n0, k0, lane);
        acc = __builtin_amdgcn_wmma_f32_16x16x32_bf16(
            false, a, false, b, (short)0, acc, false, false);
    }

    int half = lane >> 4, c = lane & 15;
    int col = n0 + c;
    float bv = bias ? bias[col] : 0.0f;
#pragma unroll
    for (int v = 0; v < 8; v++) {
        int row = m0 + v + 8 * half;
        size_t idx = (size_t)row * N + col;
        float val = acc[v] + bv;
        if (mode == MODE_GELU_BF16) {
            Cb[idx] = f2bf(gelu_exact(val));
        } else if (mode == MODE_TANHPI_F32) {
            Cf[idx] = tanhf(val) * PI_F;
        } else if (mode == MODE_RESID_F32) {
            Cf[idx] = resid[idx] + val;
        } else {
            Cf[idx] = val;
        }
    }
}

// ---------------------------------------------------------------- gate head: sigmoid(hg @ gW2 + gb2)
__global__ __launch_bounds__(256) void gate_head(const u16* __restrict__ hg,
                                                 const float* __restrict__ gW2,
                                                 const float* __restrict__ gb2,
                                                 float* __restrict__ gate) {
    int gid  = blockIdx.x * blockDim.x + threadIdx.x;
    int tok  = gid >> 5;
    int lane = gid & 31;
    if (tok >= BLTOK) return;
    float s = 0.0f;
#pragma unroll
    for (int i = lane; i < 128; i += 32) s += bf2f(hg[(size_t)tok * 128 + i]) * gW2[i];
    s = wave_sum(s);
    if (lane == 0) gate[tok] = sigm(s + gb2[0]);
}

// ---------------------------------------------------------------- phasor precompute
__global__ __launch_bounds__(256) void phasor_prep(
    const float* __restrict__ kphase, const float* __restrict__ qphase,
    const float* __restrict__ lphase, const float* __restrict__ freqs,
    float2* __restrict__ key_ph, float2* __restrict__ qry_ph,
    float2* __restrict__ ltm_ph, float2* __restrict__ jK,
    float2* __restrict__ jQ, float2* __restrict__ pos_ph)
{
    int t = blockIdx.x * blockDim.x + threadIdx.x;
    if (t < BLTOK) {
        float2 kph[NPL], qph[NPL];
#pragma unroll
        for (int j = 0; j < NPL; j++) {
            float sn, cs;
            sincosf(kphase[(size_t)t * NPL + j], &sn, &cs);
            kph[j] = make_float2(cs, sn); key_ph[(size_t)t * NPL + j] = kph[j];
            sincosf(qphase[(size_t)t * NPL + j], &sn, &cs);
            qph[j] = make_float2(cs, sn); qry_ph[(size_t)t * NPL + j] = qph[j];
            sincosf(lphase[(size_t)t * NPL + j], &sn, &cs);
            ltm_ph[(size_t)t * NPL + j] = make_float2(cs, sn);
        }
#pragma unroll
        for (int pp = 0; pp < PPSN; pp++) {
            float2 ak = make_float2(1.f, 0.f), aq = make_float2(1.f, 0.f);
#pragma unroll
            for (int s = 0; s < NSETS; s++) {
                ak = cmul(ak, kph[s * PPSN + pp]);
                aq = cmul(aq, qph[s * PPSN + pp]);
            }
            jK[(size_t)t * PPSN + pp] = ak;
            jQ[(size_t)t * PPSN + pp] = aq;
        }
    }
    if (t < LL) {
#pragma unroll
        for (int j = 0; j < NPL; j++) {
            float ang = (float)t * freqs[j] * (2.0f * PI_F);
            float sn, cs; sincosf(ang, &sn, &cs);
            pos_ph[(size_t)t * NPL + j] = make_float2(cs, sn);
        }
    }
}

// ---------------------------------------------------------------- scan 1: surprise gates
// One block per batch; thread == feature d. Shifted-cumsum prediction scans
// with cross-D reductions each step -> write_gate / ltm_gate per token.
__global__ __launch_bounds__(256) void scan_gates(
    const float2* __restrict__ jK, const float2* __restrict__ ltm_ph,
    const float* __restrict__ V, const float* __restrict__ lval,
    const float* __restrict__ ss_p, const float* __restrict__ sb_p,
    float* __restrict__ wgate, float* __restrict__ lgate)
{
    int b = blockIdx.x, tid = threadIdx.x;
    int lane = tid & 31, wv = tid >> 5;
    float ss = ss_p[0], sb = sb_p[0];

    __shared__ float2 s_jk[PPSN];
    __shared__ float2 s_lp[NPL];
    __shared__ float  sred[8][6];

    float2 aJ[PPSN], aL[NPL];
#pragma unroll
    for (int p = 0; p < PPSN; p++) aJ[p] = make_float2(0.f, 0.f);
#pragma unroll
    for (int j = 0; j < NPL; j++)  aL[j] = make_float2(0.f, 0.f);

    for (int t = 0; t < LL; t++) {
        int bt = b * LL + t;
        __syncthreads();
        if (tid < PPSN)                    s_jk[tid]     = jK[(size_t)bt * PPSN + tid];
        else if (tid >= 8 && tid < 8+NPL)  s_lp[tid - 8] = ltm_ph[(size_t)bt * NPL + (tid - 8)];
        __syncthreads();

        float v  = V[(size_t)bt * DD + tid];
        float lv = lval[(size_t)bt * DD + tid];
        if (t + 1 < LL) {
            __builtin_prefetch(&V[(size_t)(bt + 1) * DD + tid], 0, 3);
            __builtin_prefetch(&lval[(size_t)(bt + 1) * DD + tid], 0, 3);
        }

        float pred = 0.f;     // Re( sum_p acc_p * conj(jk_p) ), shifted cumsum (pre-update)
#pragma unroll
        for (int p = 0; p < PPSN; p++)
            pred += aJ[p].x * s_jk[p].x + aJ[p].y * s_jk[p].y;
        float lpred = 0.f;
#pragma unroll
        for (int j = 0; j < NPL; j++)
            lpred += aL[j].x * s_lp[j].x + aL[j].y * s_lp[j].y;

        float r[6] = { pred * pred, pred * v, v * v,
                       lpred * lpred, lpred * lv, lv * lv };
#pragma unroll
        for (int k = 0; k < 6; k++) r[k] = wave_sum(r[k]);
        if (lane == 0) {
#pragma unroll
            for (int k = 0; k < 6; k++) sred[wv][k] = r[k];
        }
        __syncthreads();
        float pn = 0, pv = 0, vn = 0, lpn = 0, lpv = 0, lvn = 0;
#pragma unroll
        for (int w = 0; w < 8; w++) {
            pn += sred[w][0]; pv += sred[w][1]; vn += sred[w][2];
            lpn += sred[w][3]; lpv += sred[w][4]; lvn += sred[w][5];
        }
        float np = sqrtf(pn), nv = sqrtf(vn);
        float fam = pv / (fmaxf(np, 1e-8f) * fmaxf(nv, 1e-8f));
        if (!(np > 1e-6f)) fam = 0.0f;
        float wg = sigm(ss * ((1.0f - fam) * 0.5f - 0.5f) + sb);

        float nlp = sqrtf(lpn), nlv = sqrtf(lvn);
        float fam2 = lpv / (fmaxf(nlp, 1e-8f) * fmaxf(nlv, 1e-8f));
        if (!(nlp > 1e-6f)) fam2 = 0.0f;
        float lg = sigm(ss * ((1.0f - fam2) * 0.5f - 0.5f) + sb);

        if (tid == 0) { wgate[bt] = wg; lgate[bt] = lg; }

        // ungated bindings accumulate AFTER prediction (shifted cumsum)
#pragma unroll
        for (int p = 0; p < PPSN; p++) {
            aJ[p].x += s_jk[p].x * v; aJ[p].y += s_jk[p].y * v;
        }
#pragma unroll
        for (int j = 0; j < NPL; j++) {
            aL[j].x += s_lp[j].x * lv; aL[j].y += s_lp[j].y * lv;
        }
    }
}

// ---------------------------------------------------------------- scan 2: gated retrieval
__global__ __launch_bounds__(256) void scan_retrieve(
    const float2* __restrict__ jK, const float2* __restrict__ jQ,
    const float2* __restrict__ kph, const float2* __restrict__ qph,
    const float2* __restrict__ pph, const float2* __restrict__ lph,
    const float* __restrict__ V, const float* __restrict__ lval,
    const float* __restrict__ wgate, const float* __restrict__ lgate,
    const float* __restrict__ gatev,
    const float* __restrict__ set_w, const float* __restrict__ pos_w,
    const float* __restrict__ ltm_w,
    float* __restrict__ hout)
{
    int b = blockIdx.x, tid = threadIdx.x;

    // softmax over set weights (4 values)
    float sw0 = set_w[0], sw1 = set_w[1], sw2 = set_w[2], sw3 = set_w[3];
    float mx = fmaxf(fmaxf(sw0, sw1), fmaxf(sw2, sw3));
    float e0 = expf(sw0 - mx), e1 = expf(sw1 - mx), e2 = expf(sw2 - mx), e3 = expf(sw3 - mx);
    float inv = 1.0f / (e0 + e1 + e2 + e3);
    float wgt[NSETS] = { e0 * inv, e1 * inv, e2 * inv, e3 * inv };
    float sig_pos = sigm(pos_w[0]);
    float sig_ltm = sigm(ltm_w[0]);

    __shared__ float2 s_jk[PPSN], s_jq[PPSN];
    __shared__ float2 s_kp[NPL], s_qp[NPL], s_pp[NPL], s_lp[NPL];
    __shared__ float  s_g[3];

    float2 aC[PPSN], aB[NPL], aP[NPL], aLg[NPL];
#pragma unroll
    for (int p = 0; p < PPSN; p++) aC[p] = make_float2(0.f, 0.f);
#pragma unroll
    for (int j = 0; j < NPL; j++) {
        aB[j] = make_float2(0.f, 0.f);
        aP[j] = make_float2(0.f, 0.f);
        aLg[j] = make_float2(0.f, 0.f);
    }

    for (int t = 0; t < LL; t++) {
        int bt = b * LL + t;
        __syncthreads();
        if (tid < 4)        s_jk[tid]      = jK[(size_t)bt * PPSN + tid];
        else if (tid < 8)   s_jq[tid - 4]  = jQ[(size_t)bt * PPSN + (tid - 4)];
        else if (tid < 24)  s_kp[tid - 8]  = kph[(size_t)bt * NPL + (tid - 8)];
        else if (tid < 40)  s_qp[tid - 24] = qph[(size_t)bt * NPL + (tid - 24)];
        else if (tid < 56)  s_pp[tid - 40] = pph[(size_t)t  * NPL + (tid - 40)];
        else if (tid < 72)  s_lp[tid - 56] = lph[(size_t)bt * NPL + (tid - 56)];
        else if (tid == 72) s_g[0] = wgate[bt];
        else if (tid == 73) s_g[1] = lgate[bt];
        else if (tid == 74) s_g[2] = gatev[bt];
        __syncthreads();

        float v  = V[(size_t)bt * DD + tid];
        float lv = lval[(size_t)bt * DD + tid];
        if (t + 1 < LL) {
            __builtin_prefetch(&V[(size_t)(bt + 1) * DD + tid], 0, 3);
            __builtin_prefetch(&lval[(size_t)(bt + 1) * DD + tid], 0, 3);
        }
        float wg = s_g[0], lg = s_g[1], g = s_g[2];
        float vg = v * wg;

        // cross-bank: cumsum incl current, retrieve with conj(joint_query)
        float cr = 0.f;
#pragma unroll
        for (int p = 0; p < PPSN; p++) {
            aC[p].x += s_jk[p].x * vg; aC[p].y += s_jk[p].y * vg;
            cr += aC[p].x * s_jq[p].x + aC[p].y * s_jq[p].y;
        }
        // per-bank
        float tot = 0.f;
#pragma unroll
        for (int s = 0; s < NSETS; s++) {
            float rs = 0.f;
#pragma unroll
            for (int pp = 0; pp < PPSN; pp++) {
                int i = s * PPSN + pp;
                aB[i].x += s_kp[i].x * vg; aB[i].y += s_kp[i].y * vg;
                rs += aB[i].x * s_qp[i].x + aB[i].y * s_qp[i].y;
            }
            tot += wgt[s] * rs;
        }
        tot = (tot + cr) * (1.0f / (NSETS + 1));

        // positional (ungated V)
        float rp = 0.f;
#pragma unroll
        for (int j = 0; j < NPL; j++) {
            aP[j].x += s_pp[j].x * v; aP[j].y += s_pp[j].y * v;
            rp += aP[j].x * s_pp[j].x + aP[j].y * s_pp[j].y;
        }
        tot = g * tot + (1.0f - g) * sig_pos * rp;

        // LTM gated
        float rl = 0.f, lvg = lv * lg;
#pragma unroll
        for (int j = 0; j < NPL; j++) {
            aLg[j].x += s_lp[j].x * lvg; aLg[j].y += s_lp[j].y * lvg;
            rl += aLg[j].x * s_lp[j].x + aLg[j].y * s_lp[j].y;
        }
        rl *= rsqrtf((float)(t + 1) * (float)NPL);
        tot += sig_ltm * rl;

        hout[(size_t)bt * DD + tid] = tot * rsqrtf((float)(t + 1) * (float)PPSN);
    }
}

// ---------------------------------------------------------------- LayerNorm -> bf16
__global__ __launch_bounds__(256) void layernorm_bf16(
    const float* __restrict__ h, const float* __restrict__ g,
    const float* __restrict__ bta, u16* __restrict__ out)
{
    int tok = blockIdx.x, tid = threadIdx.x, lane = tid & 31, wv = tid >> 5;
    __shared__ float sm[8], sq[8];
    float x = h[(size_t)tok * DD + tid];
    float s1 = wave_sum(x), s2 = wave_sum(x * x);
    if (lane == 0) { sm[wv] = s1; sq[wv] = s2; }
    __syncthreads();
    float S1 = 0, S2 = 0;
#pragma unroll
    for (int w = 0; w < 8; w++) { S1 += sm[w]; S2 += sq[w]; }
    float mu = S1 * (1.0f / DD);
    float var = S2 * (1.0f / DD) - mu * mu;
    float val = (x - mu) * rsqrtf(var + 1e-5f) * g[tid] + bta[tid];
    out[(size_t)tok * DD + tid] = f2bf(val);
}

// ---------------------------------------------------------------- launch
extern "C" void kernel_launch(void* const* d_in, const int* in_sizes, int n_in,
                              void* d_out, int out_size, void* d_ws, size_t ws_size,
                              hipStream_t stream)
{
    const float* x    = (const float*)d_in[0];
    const float* kW1  = (const float*)d_in[1];  const float* kb1 = (const float*)d_in[2];
    const float* kW2  = (const float*)d_in[3];  const float* kb2 = (const float*)d_in[4];
    const float* qW1  = (const float*)d_in[5];  const float* qb1 = (const float*)d_in[6];
    const float* qW2  = (const float*)d_in[7];  const float* qb2 = (const float*)d_in[8];
    const float* vW   = (const float*)d_in[9];  const float* vb  = (const float*)d_in[10];
    const float* ln_g = (const float*)d_in[11]; const float* ln_b = (const float*)d_in[12];
    const float* oW   = (const float*)d_in[13]; const float* ob  = (const float*)d_in[14];
    const float* set_w = (const float*)d_in[15];
    const float* pos_freqs = (const float*)d_in[16];
    const float* pos_weight = (const float*)d_in[17];
    const float* gW1  = (const float*)d_in[18]; const float* gb1 = (const float*)d_in[19];
    const float* gW2  = (const float*)d_in[20]; const float* gb2 = (const float*)d_in[21];
    const float* lkW1 = (const float*)d_in[22]; const float* lkb1 = (const float*)d_in[23];
    const float* lkW2 = (const float*)d_in[24]; const float* lkb2 = (const float*)d_in[25];
    const float* lvW  = (const float*)d_in[26]; const float* lvb = (const float*)d_in[27];
    const float* ss_p = (const float*)d_in[28];
    const float* sb_p = (const float*)d_in[29];
    const float* lw_p = (const float*)d_in[30];

    // ---- workspace carve-up
    char* wsb = (char*)d_ws; size_t off = 0;
    auto alloc = [&](size_t bytes) -> void* {
        void* p = wsb + off; off = (off + bytes + 255) & ~(size_t)255; return p;
    };
    u16* x_bf   = (u16*)alloc((size_t)BLTOK * DD * 2);
    u16* kW1t   = (u16*)alloc((size_t)DD * DD * 2);
    u16* qW1t   = (u16*)alloc((size_t)DD * DD * 2);
    u16* vWt    = (u16*)alloc((size_t)DD * DD * 2);
    u16* lkW1t  = (u16*)alloc((size_t)DD * DD * 2);
    u16* lvWt   = (u16*)alloc((size_t)DD * DD * 2);
    u16* oWt    = (u16*)alloc((size_t)DD * DD * 2);
    u16* gW1t   = (u16*)alloc((size_t)128 * DD * 2);
    u16* kW2t   = (u16*)alloc((size_t)NPL * DD * 2);
    u16* qW2t   = (u16*)alloc((size_t)NPL * DD * 2);
    u16* lkW2t  = (u16*)alloc((size_t)NPL * DD * 2);
    u16* hk     = (u16*)alloc((size_t)BLTOK * DD * 2);
    u16* hq     = (u16*)alloc((size_t)BLTOK * DD * 2);
    u16* hl     = (u16*)alloc((size_t)BLTOK * DD * 2);
    u16* hg     = (u16*)alloc((size_t)BLTOK * 128 * 2);
    float* Vbuf = (float*)alloc((size_t)BLTOK * DD * 4);
    float* lvalb= (float*)alloc((size_t)BLTOK * DD * 4);
    float* kphase = (float*)alloc((size_t)BLTOK * NPL * 4);
    float* qphase = (float*)alloc((size_t)BLTOK * NPL * 4);
    float* lphase = (float*)alloc((size_t)BLTOK * NPL * 4);
    float* gate   = (float*)alloc((size_t)BLTOK * 4);
    float* wgate  = (float*)alloc((size_t)BLTOK * 4);
    float* lgate  = (float*)alloc((size_t)BLTOK * 4);
    float2* key_ph = (float2*)alloc((size_t)BLTOK * NPL * 8);
    float2* qry_ph = (float2*)alloc((size_t)BLTOK * NPL * 8);
    float2* ltm_ph = (float2*)alloc((size_t)BLTOK * NPL * 8);
    float2* jK     = (float2*)alloc((size_t)BLTOK * PPSN * 8);
    float2* jQ     = (float2*)alloc((size_t)BLTOK * PPSN * 8);
    float2* pos_ph = (float2*)alloc((size_t)LL * NPL * 8);
    float* hbuf    = (float*)alloc((size_t)BLTOK * DD * 4);
    u16* hn_bf     = (u16*)alloc((size_t)BLTOK * DD * 2);

    // ---- prep
    {
        int n = BLTOK * DD;
        cast_f32_to_bf16<<<(n + 255) / 256, 256, 0, stream>>>(x, x_bf, n);
    }
    auto tc = [&](const float* W, u16* Wt, int din, int dout) {
        int n = din * dout;
        transpose_cast<<<(n + 255) / 256, 256, 0, stream>>>(W, Wt, din, dout);
    };
    tc(kW1, kW1t, DD, DD);   tc(qW1, qW1t, DD, DD);
    tc(vW, vWt, DD, DD);     tc(lkW1, lkW1t, DD, DD);
    tc(lvW, lvWt, DD, DD);   tc(oW, oWt, DD, DD);
    tc(gW1, gW1t, DD, 128);
    tc(kW2, kW2t, DD, NPL);  tc(qW2, qW2t, DD, NPL);  tc(lkW2, lkW2t, DD, NPL);

    // ---- stage-1 projections (WMMA)
    dim3 blk(128);
    auto gemm = [&](const u16* A, const u16* Bt, const float* bias, const float* resid,
                    float* Cf, u16* Cb, int M, int N, int K, int mode) {
        dim3 grid(M / 16, (N + 63) / 64);
        wmma_gemm<<<grid, blk, 0, stream>>>(A, Bt, bias, resid, Cf, Cb, M, N, K, mode);
    };
    gemm(x_bf, kW1t,  kb1,  nullptr, nullptr, hk, BLTOK, DD,  DD, MODE_GELU_BF16);
    gemm(x_bf, qW1t,  qb1,  nullptr, nullptr, hq, BLTOK, DD,  DD, MODE_GELU_BF16);
    gemm(x_bf, lkW1t, lkb1, nullptr, nullptr, hl, BLTOK, DD,  DD, MODE_GELU_BF16);
    gemm(x_bf, gW1t,  gb1,  nullptr, nullptr, hg, BLTOK, 128, DD, MODE_GELU_BF16);
    gemm(x_bf, vWt,   vb,   nullptr, Vbuf,  nullptr, BLTOK, DD, DD, MODE_F32);
    gemm(x_bf, lvWt,  lvb,  nullptr, lvalb, nullptr, BLTOK, DD, DD, MODE_F32);

    // ---- stage-2 phase heads (WMMA) + gate head
    gemm(hk, kW2t,  kb2,  nullptr, kphase, nullptr, BLTOK, NPL, DD, MODE_TANHPI_F32);
    gemm(hq, qW2t,  qb2,  nullptr, qphase, nullptr, BLTOK, NPL, DD, MODE_TANHPI_F32);
    gemm(hl, lkW2t, lkb2, nullptr, lphase, nullptr, BLTOK, NPL, DD, MODE_TANHPI_F32);
    gate_head<<<(BLTOK * 32) / 256, 256, 0, stream>>>(hg, gW2, gb2, gate);

    // ---- phasors
    phasor_prep<<<(BLTOK + 255) / 256, 256, 0, stream>>>(
        kphase, qphase, lphase, pos_freqs,
        key_ph, qry_ph, ltm_ph, jK, jQ, pos_ph);

    // ---- sequential scans
    scan_gates<<<BB, 256, 0, stream>>>(jK, ltm_ph, Vbuf, lvalb, ss_p, sb_p, wgate, lgate);
    scan_retrieve<<<BB, 256, 0, stream>>>(jK, jQ, key_ph, qry_ph, pos_ph, ltm_ph,
                                          Vbuf, lvalb, wgate, lgate, gate,
                                          set_w, pos_weight, lw_p, hbuf);

    // ---- LayerNorm + output projection + residual (WMMA)
    layernorm_bf16<<<BLTOK, 256, 0, stream>>>(hbuf, ln_g, ln_b, hn_bf);
    gemm(hn_bf, oWt, ob, x, (float*)d_out, nullptr, BLTOK, DD, DD, MODE_RESID_F32);
}
